// MultiHeadAttentionLayer_4896262717532
// MI455X (gfx1250) — matrix operand order repaired and checked by reading, not compiled, not run on hardware
//
#include <hip/hip_runtime.h>

// Problem sizes (fixed by the reference)
#define HIDDEN 1024
#define S_LEN  2048
#define BATCH  4
#define HEADS  16
#define HDIM   64
#define ROWS   (BATCH * S_LEN)   // 8192

typedef __attribute__((ext_vector_type(16))) __bf16 v16bf;
typedef __attribute__((ext_vector_type(8)))  float  v8f;
typedef __attribute__((ext_vector_type(4)))  unsigned int v4u;

union Frag { v16bf v; v4u q[2]; unsigned int u[8]; };

__device__ __forceinline__ unsigned short f2bf(float f) {
  unsigned int x = __float_as_uint(f);
  x += 0x7FFFu + ((x >> 16) & 1u);   // round-to-nearest-even
  return (unsigned short)(x >> 16);
}
__device__ __forceinline__ unsigned int pack2bf(float lo, float hi) {
  return (unsigned int)f2bf(lo) | ((unsigned int)f2bf(hi) << 16);
}

// WMMA bf16 A-layout (ISA 7.12.2): lane (m, half g) VGPR j holds K-pair
//   j<4 : kp = g*4 + j ; j>=4 : kp = 8 + g*4 + (j-4)
// Inverse (tile writers): given kp: g = (kp&7)>>2, j = (kp>>3)*4 + (kp&3)
// Fragment groups stored as 8 consecutive dwords (j order), padded to 48 B
// (3 x uint4) so every fragment read is two aligned ds_load_b128.
#define GRP 12   // dwords per (row, half) group incl. 4-dword pad

// ---------------------------------------------------------------------------
// Stage 1: Y = X @ W + b  (8192x1024 @ 1024x1024), output bf16 to workspace.
// ---------------------------------------------------------------------------
__global__ __launch_bounds__(256)
void proj_kernel(const float* __restrict__ q_in, const float* __restrict__ k_in,
                 const float* __restrict__ v_in,
                 const float* __restrict__ q_w, const float* __restrict__ k_w,
                 const float* __restrict__ v_w,
                 const float* __restrict__ q_b, const float* __restrict__ k_b,
                 const float* __restrict__ v_b,
                 unsigned short* __restrict__ q_ws,
                 unsigned short* __restrict__ k_ws,
                 unsigned short* __restrict__ v_ws) {
  const float* X; const float* W; const float* Bv; unsigned short* Y;
  if (blockIdx.z == 0)      { X = q_in; W = q_w; Bv = q_b; Y = q_ws; }
  else if (blockIdx.z == 1) { X = k_in; W = k_w; Bv = k_b; Y = k_ws; }
  else                      { X = v_in; W = v_w; Bv = v_b; Y = v_ws; }

  __shared__ v4u AfrQ[128 * 2 * 3];   // A frags: (m, g) groups, 12 KB
  __shared__ v4u BfrQ[64 * 2 * 3];    // B frags: (n, half) groups, 6 KB
  unsigned int* Afr = (unsigned int*)AfrQ;
  unsigned int* Bfr = (unsigned int*)BfrQ;

  const int tid  = threadIdx.x;
  const int lane = tid & 31;
  const int wave = tid >> 5;
  const int half = lane >> 4;
  const int l15  = lane & 15;
  const int m0   = blockIdx.y * 128;
  const int n0   = blockIdx.x * 64;

  v8f acc[4] = {};

  for (int k0 = 0; k0 < HIDDEN; k0 += 32) {
    if (k0 + 32 < HIDDEN) {   // prefetch next k-step tiles (global_prefetch_b8)
      __builtin_prefetch(X + (size_t)(m0 + (tid >> 4)) * HIDDEN + k0 + 32 + 2 * (tid & 15), 0, 1);
      __builtin_prefetch(W + (size_t)(k0 + 32 + (tid >> 6) * 8) * HIDDEN + n0 + (tid & 63), 0, 1);
    }
    __syncthreads();
    // A tile: 2048 dwords; contiguous-k float pairs -> bf16 pair dword
#pragma unroll
    for (int i = 0; i < 8; i++) {
      int idx = tid + i * 256;
      int m = idx >> 4, kp = idx & 15;
      int g = (kp & 7) >> 2, j = (kp >> 3) * 4 + (kp & 3);
      const float* p = X + (size_t)(m0 + m) * HIDDEN + k0 + 2 * kp;
      Afr[(m * 2 + g) * GRP + j] = pack2bf(p[0], p[1]);
    }
    // B tile: 1024 dwords; pair spans two consecutive k-rows of W
#pragma unroll
    for (int i = 0; i < 4; i++) {
      int idx = tid + i * 256;
      int kpb = (idx >> 6) & 15, n = idx & 63;
      int bh = kpb >> 3, bj = kpb & 7;
      const float* p = W + (size_t)(k0 + (bj >> 2) * 16 + bh * 8 + 2 * (bj & 3)) * HIDDEN + n0 + n;
      Bfr[(n * 2 + bh) * GRP + bj] = pack2bf(p[0], p[HIDDEN]);
    }
    __syncthreads();

    // Load A + all 4 B fragments first, then 4 back-to-back WMMAs.
    Frag af, bfr[4];
    const int abase = ((wave * 16 + l15) * 2 + half) * 3;
    af.q[0] = AfrQ[abase];
    af.q[1] = AfrQ[abase + 1];
#pragma unroll
    for (int nt = 0; nt < 4; nt++) {
      const int bbase = ((nt * 16 + l15) * 2 + half) * 3;
      bfr[nt].q[0] = BfrQ[bbase];
      bfr[nt].q[1] = BfrQ[bbase + 1];
    }
#pragma unroll
    for (int nt = 0; nt < 4; nt++)
      acc[nt] = __builtin_amdgcn_wmma_f32_16x16x32_bf16(
          false, af.v, false, bfr[nt].v, (short)0, acc[nt], false, false);
  }

  // D layout: VGPR r, lane -> row = r + 8*half, col = l15 (per 16x16 tile)
#pragma unroll
  for (int nt = 0; nt < 4; nt++) {
#pragma unroll
    for (int r = 0; r < 8; r++) {
      int row = m0 + wave * 16 + r + 8 * half;
      int col = n0 + nt * 16 + l15;
      Y[(size_t)row * HIDDEN + col] = f2bf(acc[nt][r] + Bv[col]);
    }
  }
}

// ---------------------------------------------------------------------------
// Stage 2: per (b,h):  M = K_h^T V_h  (64x64), then ctx = scale * Q_h @ M.
// Exploits linearity of the (bug-faithful) softmax-free attention.
// ---------------------------------------------------------------------------
__global__ __launch_bounds__(256)
void attn_kernel(const unsigned short* __restrict__ Qws,
                 const unsigned short* __restrict__ Kws,
                 const unsigned short* __restrict__ Vws,
                 float* __restrict__ out) {
  const int bh = blockIdx.x;
  const int b = bh >> 4, h = bh & 15;
  const int hbase = h * HDIM;
  const size_t rowbase = (size_t)b * S_LEN;

  // One 56 KB LDS block, manually laid out / phase-aliased:
  //   [0      .. 12288): K^T A-frags (phase A)  -> M B-frags (phase B prep)
  //   [12288  .. 24576): V  B-frags (phase A)
  //   [24576  .. 57344): Mpart f32 (phase A reduce)
  //   phase B: whole block reused as 8 wave-private double-buffered Q tiles
  //            (wave w at w*4608, 2 x 2304 B, rows padded to 144 B)
  __shared__ v4u smem[3584];                  // 57,344 B
  v4u*  KfrQ  = smem;                         // 768 v4u
  v4u*  VfrQ  = smem + 768;                   // 768 v4u
  float* Mpart = (float*)(smem + 1536);       // 8192 f32
  v4u*  MfrQ  = smem;                         // alias: Kfr dead after phase A
  unsigned int* Kfr = (unsigned int*)KfrQ;
  unsigned int* Vfr = (unsigned int*)VfrQ;
  unsigned int* Mfr = (unsigned int*)MfrQ;

  const int tid  = threadIdx.x;
  const int lane = tid & 31;
  const int wave = tid >> 5;
  const int half = lane >> 4;
  const int l15  = lane & 15;
  const int mtile = wave & 3;   // which 16 rows of M this wave produces
  const int shalf = wave >> 2;  // which half of S this wave accumulates

  // ---- Phase A: M = K^T V, split over 2 S-halves x 4 m-tiles ----
  v8f macc[4] = {};
  for (int step = 0; step < 32; step++) {
    __syncthreads();
    // K^T tile as A-fragments: 2048 dwords (pair spans 2 consecutive s-rows)
#pragma unroll
    for (int i = 0; i < 8; i++) {
      int idx = tid + i * 256;
      int sh = idx >> 10, m = idx & 63, kp = (idx >> 6) & 15;
      int g = (kp & 7) >> 2, j = (kp >> 3) * 4 + (kp & 3);
      int s = sh * 1024 + step * 32 + 2 * kp;
      const unsigned short* p = Kws + (rowbase + s) * HIDDEN + hbase + m;
      Kfr[sh * (64 * 2 * GRP) + (m * 2 + g) * GRP + j] =
          (unsigned int)p[0] | ((unsigned int)p[HIDDEN] << 16);
    }
    // V tile as B-fragments: 2048 dwords
#pragma unroll
    for (int i = 0; i < 8; i++) {
      int idx = tid + i * 256;
      int sh = idx >> 10, n = idx & 63, kpb = (idx >> 6) & 15;
      int bh2 = kpb >> 3, bj = kpb & 7;
      int s = sh * 1024 + step * 32 + (bj >> 2) * 16 + bh2 * 8 + 2 * (bj & 3);
      const unsigned short* p = Vws + (rowbase + s) * HIDDEN + hbase + n;
      Vfr[sh * (64 * 2 * GRP) + (n * 2 + bh2) * GRP + bj] =
          (unsigned int)p[0] | ((unsigned int)p[HIDDEN] << 16);
    }
    __syncthreads();

    Frag af, bfr[4];
    const int abase = shalf * (64 * 2 * 3) + ((mtile * 16 + l15) * 2 + half) * 3;
    af.q[0] = KfrQ[abase];
    af.q[1] = KfrQ[abase + 1];
#pragma unroll
    for (int nt = 0; nt < 4; nt++) {
      const int bbase = shalf * (64 * 2 * 3) + ((nt * 16 + l15) * 2 + half) * 3;
      bfr[nt].q[0] = VfrQ[bbase];
      bfr[nt].q[1] = VfrQ[bbase + 1];
    }
#pragma unroll
    for (int nt = 0; nt < 4; nt++)
      macc[nt] = __builtin_amdgcn_wmma_f32_16x16x32_bf16(
          false, af.v, false, bfr[nt].v, (short)0, macc[nt], false, false);
  }

  __syncthreads();
#pragma unroll
  for (int nt = 0; nt < 4; nt++)
#pragma unroll
    for (int r = 0; r < 8; r++)
      Mpart[(shalf * 64 + mtile * 16 + r + 8 * half) * 64 + nt * 16 + l15] =
          macc[nt][r];
  __syncthreads();

  // Reduce S-halves, scale, convert M to bf16 B-fragment groups (aliases Kfr).
  const float scale = 0.125f;  // 1/sqrt(64)
#pragma unroll
  for (int i = 0; i < 8; i++) {
    int idx = tid + i * 256;
    int kt = idx >> 10, n = idx & 63, kpb = (idx >> 6) & 15;
    int bh2 = kpb >> 3, bj = kpb & 7;
    int kk = kt * 32 + (bj >> 2) * 16 + bh2 * 8 + 2 * (bj & 3);
    float lo = scale * (Mpart[kk * 64 + n]       + Mpart[4096 + kk * 64 + n]);
    float hi = scale * (Mpart[(kk + 1) * 64 + n] + Mpart[4096 + (kk + 1) * 64 + n]);
    Mfr[kt * (64 * 2 * GRP) + (n * 2 + bh2) * GRP + bj] = pack2bf(lo, hi);
  }
  __syncthreads();

  // Hold all 8 B-fragments of M in registers for the whole phase B.
  Frag mb[2][4];
#pragma unroll
  for (int kt = 0; kt < 2; kt++)
#pragma unroll
    for (int nt = 0; nt < 4; nt++) {
      const int base = kt * (64 * 2 * 3) + ((nt * 16 + l15) * 2 + half) * 3;
      mb[kt][nt].q[0] = MfrQ[base];
      mb[kt][nt].q[1] = MfrQ[base + 1];
    }
  __syncthreads();   // Mfr region about to be recycled as Q buffers

  // ---- Phase B: ctx = Q @ M, double-buffered via async DMA to LDS.
  // Each wave streams its 16x64 bf16 Q tile (2 KB) with 4x
  // global_load_async_to_lds_b128 (ASYNCcnt), overlapping tile t+1's copy
  // with tile t's WMMAs. Buffers are wave-private; rows padded to 144 B.
  unsigned char* qb_base = (unsigned char*)smem + (size_t)wave * 4608;

  auto issue_tile = [&](int t, int buf) {
#pragma unroll
    for (int i = 0; i < 4; i++) {
      int fl = lane + 32 * i;
      int r = fl >> 3, c = fl & 7;              // row 0..15, 16B chunk 0..7
      int s = wave * 256 + t * 16 + r;
      const void* g = (const unsigned char*)Qws +
                      ((rowbase + s) * HIDDEN + hbase) * 2 + c * 16;
      unsigned lds = (unsigned)(size_t)(qb_base + buf * 2304 + r * 144 + c * 16);
      asm volatile("global_load_async_to_lds_b128 %0, %1, off"
                   :: "v"(lds), "v"((unsigned long long)g)
                   : "memory");
    }
  };

  issue_tile(0, 0);
  for (int t = 0; t < 16; t++) {
    const int cur = t & 1;
    if (t + 1 < 16) {
      issue_tile(t + 1, cur ^ 1);
      asm volatile("s_wait_asynccnt 0x4" ::: "memory");  // tile t resident
    } else {
      asm volatile("s_wait_asynccnt 0x0" ::: "memory");
    }

    const v4u* qb = (const v4u*)(qb_base + cur * 2304);
    v8f cacc[4] = {};
#pragma unroll
    for (int kt = 0; kt < 2; kt++) {
      Frag af;  // two contiguous 16B runs of this lane's Q row
      af.q[0] = qb[l15 * 9 + kt * 4 + half];
      af.q[1] = qb[l15 * 9 + kt * 4 + half + 2];
#pragma unroll
      for (int nt = 0; nt < 4; nt++)
        cacc[nt] = __builtin_amdgcn_wmma_f32_16x16x32_bf16(
            false, af.v, false, mb[kt][nt].v, (short)0, cacc[nt], false, false);
    }

#pragma unroll
    for (int nt = 0; nt < 4; nt++)
#pragma unroll
      for (int r = 0; r < 8; r++) {
        int s = wave * 256 + t * 16 + r + 8 * half;
        out[(rowbase + s) * HIDDEN + hbase + nt * 16 + l15] = cacc[nt][r];
      }
  }
}

// ---------------------------------------------------------------------------
extern "C" void kernel_launch(void* const* d_in, const int* in_sizes, int n_in,
                              void* d_out, int out_size, void* d_ws, size_t ws_size,
                              hipStream_t stream) {
  const float* q  = (const float*)d_in[0];
  const float* k  = (const float*)d_in[1];
  const float* v  = (const float*)d_in[2];
  const float* qw = (const float*)d_in[3];
  const float* qb = (const float*)d_in[4];
  const float* kw = (const float*)d_in[5];
  const float* kb = (const float*)d_in[6];
  const float* vw = (const float*)d_in[7];
  const float* vb = (const float*)d_in[8];

  unsigned short* Qws = (unsigned short*)d_ws;          // 16 MB bf16 each
  unsigned short* Kws = Qws + (size_t)ROWS * HIDDEN;
  unsigned short* Vws = Kws + (size_t)ROWS * HIDDEN;
  float* out = (float*)d_out;

  dim3 g1(HIDDEN / 64, ROWS / 128, 3);                  // 16 x 64 x 3 blocks
  proj_kernel<<<g1, 256, 0, stream>>>(q, k, v, qw, kw, vw, qb, kb, vb,
                                      Qws, Kws, Vws);
  attn_kernel<<<dim3(BATCH * HEADS), 256, 0, stream>>>(Qws, Kws, Vws, out);
}
